// BiGSLayer_68822555951839
// MI455X (gfx1250) — matrix operand order repaired and checked by reading, not compiled, not run on hardware
//
#include <hip/hip_runtime.h>
#include <hip/hip_bf16.h>

typedef __bf16 bf16;
typedef __attribute__((ext_vector_type(16))) __bf16 v16bf;
typedef __attribute__((ext_vector_type(8)))  __bf16 bf16x8;
typedef __attribute__((ext_vector_type(8)))  float  v8f;

#define HIDDEN   1024
#define INTER    512
#define D_STATE  16
#define D_CONV   4
#define D_INNER  2048
#define DT_RANK  64
#define BATCH    4
#define SEQ      2048
#define MTOT     (BATCH * SEQ)

#define ACT_NONE     0
#define ACT_GELU     1
#define ACT_SILU     2
#define ACT_SOFTPLUS 3

// ---------------------------------------------------------------- helpers
template <int ACT>
__device__ __forceinline__ float act_apply(float x) {
  if (ACT == ACT_GELU)     return 0.5f * x * (1.0f + erff(x * 0.70710678118654752f));
  if (ACT == ACT_SILU)     return x / (1.0f + __expf(-x));
  if (ACT == ACT_SOFTPLUS) return (x > 20.0f) ? x : log1pf(__expf(x));
  return x;
}

// ---------------------------------------------------------------- f32 -> bf16
__global__ __launch_bounds__(256) void cvt_f32_bf16(const float* __restrict__ in,
                                                    bf16* __restrict__ out, int n) {
  int i = blockIdx.x * 256 + threadIdx.x;
  if (i < n) out[i] = (bf16)in[i];
}

// ---------------------------------------------------------------- LayerNorm
__global__ __launch_bounds__(256) void ln_kernel(const float* __restrict__ x,
                                                 const float* __restrict__ g,
                                                 const float* __restrict__ b,
                                                 bf16* __restrict__ y) {
  __shared__ float red[256];
  const int row = blockIdx.x;
  const float* xr = x + (size_t)row * HIDDEN;
  float s = 0.f;
  for (int i = threadIdx.x; i < HIDDEN; i += 256) s += xr[i];
  red[threadIdx.x] = s; __syncthreads();
  for (int o = 128; o > 0; o >>= 1) {
    if (threadIdx.x < o) red[threadIdx.x] += red[threadIdx.x + o];
    __syncthreads();
  }
  const float mu = red[0] * (1.0f / HIDDEN);
  __syncthreads();
  float v = 0.f;
  for (int i = threadIdx.x; i < HIDDEN; i += 256) { float d = xr[i] - mu; v += d * d; }
  red[threadIdx.x] = v; __syncthreads();
  for (int o = 128; o > 0; o >>= 1) {
    if (threadIdx.x < o) red[threadIdx.x] += red[threadIdx.x + o];
    __syncthreads();
  }
  const float inv = rsqrtf(red[0] * (1.0f / HIDDEN) + 1e-6f);
  for (int i = threadIdx.x; i < HIDDEN; i += 256)
    y[(size_t)row * HIDDEN + i] = (bf16)((xr[i] - mu) * inv * g[i] + b[i]);
}

// ---------------------------------------------------------------- WMMA GEMM
// C[M,N] = act(A[M,K] @ W[N,K]^T + bias) (* mul) (+ residual), optional seq-flip of out rows.
// Block tile 128x128, 8 waves -> wave tile 64x32 (4x2 WMMA accums), double-buffered LDS,
// one barrier per k-tile, global->reg prefetch of the next tile.
#define BM 128
#define BN 128
#define BK 32
#define LDSS 40   // padded LDS row stride (elements) to dodge bank conflicts

template <int ACT, bool FLIP, bool HASMUL, bool HASRES, bool OUTF32>
__global__ __launch_bounds__(256) void gemm_bf16_kernel(
    const bf16* __restrict__ A, int lda,
    const bf16* __restrict__ W,             // [N,K] row-major (ldb == K)
    const float* __restrict__ bias,         // nullable, [N]
    const bf16* __restrict__ mul,           // [M,N] indexed at FINAL row (if HASMUL)
    const float* __restrict__ residual,     // [M,N] indexed at FINAL row (if HASRES)
    bf16* __restrict__ outb, float* __restrict__ outf,
    int N, int K) {
  __shared__ bf16 As[2][BM * LDSS];
  __shared__ bf16 Ws[2][BN * LDSS];

  const int tid  = threadIdx.x;
  const int lane = tid & 31;
  const int wave = tid >> 5;
  const int wm   = wave >> 2;   // 0..1  -> 64 rows each
  const int wn   = wave & 3;    // 0..3  -> 32 cols each
  const int m0   = blockIdx.y * BM;
  const int n0   = blockIdx.x * BN;

  const int laneRow = lane & 15;
  const int laneHi  = lane >> 4;

  // staging: each thread moves 16 A elems + 16 W elems per k-tile
  const int srow = tid >> 1;            // 0..127
  const int scol = (tid & 1) * 16;      // 0 / 16
  const bf16* aptr = A + (size_t)(m0 + srow) * lda + scol;
  const int nclamp = (n0 + srow < N) ? (n0 + srow) : (N - 1);  // epilogue discards ncol>=N
  const bf16* wptr = W + (size_t)nclamp * K + scol;

  v8f acc[4][2];
#pragma unroll
  for (int i = 0; i < 4; ++i)
#pragma unroll
    for (int j = 0; j < 2; ++j)
#pragma unroll
      for (int e = 0; e < 8; ++e) acc[i][j][e] = 0.0f;

  // prologue: tile 0 -> buffer 0
  {
    bf16x8 a0 = *(const bf16x8*)aptr;
    bf16x8 a1 = *(const bf16x8*)(aptr + 8);
    bf16x8 w0 = *(const bf16x8*)wptr;
    bf16x8 w1 = *(const bf16x8*)(wptr + 8);
    *(bf16x8*)&As[0][srow * LDSS + scol]     = a0;
    *(bf16x8*)&As[0][srow * LDSS + scol + 8] = a1;
    *(bf16x8*)&Ws[0][srow * LDSS + scol]     = w0;
    *(bf16x8*)&Ws[0][srow * LDSS + scol + 8] = w1;
  }
  __syncthreads();

  const int nt = K / BK;
  for (int kt = 0; kt < nt; ++kt) {
    const int cur = kt & 1;
    const bool more = (kt + 1 < nt);

    bf16x8 na0, na1, nw0, nw1;
    if (more) {
      const bf16* ap = aptr + (kt + 1) * BK;
      const bf16* wp = wptr + (kt + 1) * BK;
      na0 = *(const bf16x8*)ap;
      na1 = *(const bf16x8*)(ap + 8);
      nw0 = *(const bf16x8*)wp;
      nw1 = *(const bf16x8*)(wp + 8);
      if (kt + 2 < nt) __builtin_prefetch((const void*)(aptr + (kt + 2) * BK), 0, 0);
    }

    // fragments per ISA 16-bit layouts (A 16x32, B 32x16)
    v16bf afrag[4], bfrag[2];
#pragma unroll
    for (int i = 0; i < 4; ++i) {
      const int r  = wm * 64 + i * 16 + laneRow;
      const int kb = laneHi * 8;
      bf16x8 c0 = *(const bf16x8*)&As[cur][r * LDSS + kb];
      bf16x8 c1 = *(const bf16x8*)&As[cur][r * LDSS + kb + 16];
#pragma unroll
      for (int e = 0; e < 8; ++e) { afrag[i][e] = c0[e]; afrag[i][8 + e] = c1[e]; }
    }
#pragma unroll
    for (int j = 0; j < 2; ++j) {
      const int r  = wn * 32 + j * 16 + laneRow;
      const int kb = laneHi * 8;
      bf16x8 c0 = *(const bf16x8*)&Ws[cur][r * LDSS + kb];
      bf16x8 c1 = *(const bf16x8*)&Ws[cur][r * LDSS + kb + 16];
#pragma unroll
      for (int e = 0; e < 8; ++e) { bfrag[j][e] = c0[e]; bfrag[j][8 + e] = c1[e]; }
    }

#pragma unroll
    for (int i = 0; i < 4; ++i)
#pragma unroll
      for (int j = 0; j < 2; ++j)
        acc[i][j] = __builtin_amdgcn_wmma_f32_16x16x32_bf16(
            false, afrag[i], false, bfrag[j], (short)0, acc[i][j], false, false);

    if (more) {
      const int nxt = cur ^ 1;
      *(bf16x8*)&As[nxt][srow * LDSS + scol]     = na0;
      *(bf16x8*)&As[nxt][srow * LDSS + scol + 8] = na1;
      *(bf16x8*)&Ws[nxt][srow * LDSS + scol]     = nw0;
      *(bf16x8*)&Ws[nxt][srow * LDSS + scol + 8] = nw1;
      __syncthreads();
    }
  }

  // fused epilogue: bias -> act -> flip -> mul -> residual -> store
#pragma unroll
  for (int i = 0; i < 4; ++i) {
#pragma unroll
    for (int j = 0; j < 2; ++j) {
      const int ncol = n0 + wn * 32 + j * 16 + laneRow;
      if (ncol >= N) continue;
      const float bval = bias ? bias[ncol] : 0.0f;
#pragma unroll
      for (int r = 0; r < 8; ++r) {
        const int m = m0 + wm * 64 + i * 16 + r + laneHi * 8;
        float x = acc[i][j][r] + bval;
        x = act_apply<ACT>(x);
        int mf = m;
        if (FLIP) { const int bb = m / SEQ, ss = m % SEQ; mf = bb * SEQ + (SEQ - 1 - ss); }
        const size_t off = (size_t)mf * N + ncol;
        if (HASMUL) x *= (float)mul[off];
        if (HASRES) x += residual[off];
        if (OUTF32) outf[off] = x;
        else        outb[off] = (bf16)x;
      }
    }
  }
}

// ---------------------------------------------------------------- depthwise causal conv + bias + SiLU
__global__ __launch_bounds__(256) void conv_silu_kernel(const bf16* __restrict__ xz,
                                                        const float* __restrict__ cw,
                                                        const float* __restrict__ cb,
                                                        bf16* __restrict__ out) {
  const int i = blockIdx.x * 256 + threadIdx.x;
  const int total = BATCH * SEQ * D_INNER;
  if (i >= total) return;
  const int d  = i % D_INNER;
  const int bs = i / D_INNER;
  const int s  = bs % SEQ;
  const int b  = bs / SEQ;
  float acc = cb[d];
#pragma unroll
  for (int j = 0; j < D_CONV; ++j) {
    const int ss = s - (D_CONV - 1) + j;
    if (ss >= 0)
      acc += cw[d * D_CONV + j] * (float)xz[(size_t)(b * SEQ + ss) * (2 * D_INNER) + d];
  }
  const float y = acc / (1.0f + __expf(-acc));
  out[(size_t)bs * D_INNER + d] = (bf16)y;
}

// ---------------------------------------------------------------- selective scan (+ D skip + z-gate)
__global__ __launch_bounds__(256) void scan_kernel(const bf16* __restrict__ delta,  // [B*S, D_INNER]
                                                   const bf16* __restrict__ u,      // [B*S, D_INNER]
                                                   const bf16* __restrict__ xdbl,   // [B*S, 96]
                                                   const bf16* __restrict__ xz,     // [B*S, 2*D_INNER]
                                                   const float* __restrict__ A_log, // [D_INNER, 16]
                                                   const float* __restrict__ Dp,    // [D_INNER]
                                                   bf16* __restrict__ y) {          // [B*S, D_INNER]
  const int d = blockIdx.x * 256 + threadIdx.x;
  const int b = blockIdx.y;
  float Ar[D_STATE], h[D_STATE];
#pragma unroll
  for (int n = 0; n < D_STATE; ++n) { Ar[n] = -__expf(A_log[d * D_STATE + n]); h[n] = 0.0f; }
  const float Dd = Dp[d];
  for (int s = 0; s < SEQ; ++s) {
    const size_t row = (size_t)b * SEQ + s;
    const float dt = (float)delta[row * D_INNER + d];
    const float ut = (float)u[row * D_INNER + d];
    const float du = dt * ut;
    // B (cols 64..79) and C (cols 80..95) are 64 contiguous bytes: 4x b128 loads
    const bf16x8* xv = (const bf16x8*)(xdbl + row * 96 + DT_RANK);
    const bf16x8 bv0 = xv[0], bv1 = xv[1], cv0 = xv[2], cv1 = xv[3];
    float yv = 0.0f;
#pragma unroll
    for (int n = 0; n < 8; ++n) {
      h[n] = h[n] * __expf(dt * Ar[n]) + du * (float)bv0[n];
      yv += h[n] * (float)cv0[n];
    }
#pragma unroll
    for (int n = 0; n < 8; ++n) {
      h[8 + n] = h[8 + n] * __expf(dt * Ar[8 + n]) + du * (float)bv1[n];
      yv += h[8 + n] * (float)cv1[n];
    }
    yv += ut * Dd;
    const float z = (float)xz[row * (2 * D_INNER) + D_INNER + d];
    const float g = z / (1.0f + __expf(-z));
    y[row * D_INNER + d] = (bf16)(yv * g);
  }
}

// ---------------------------------------------------------------- input order (setup_inputs dict order, DFS)
enum {
  IN_HS = 0,
  IN_LN_G, IN_LN_B,
  IN_DV_W, IN_DV_B,
  IN_DUF_W, IN_DUF_B,
  IN_DUB_W, IN_DUB_B,
  IN_DUCF_W, IN_DUCF_B,
  IN_DUCB_W, IN_DUCB_B,
  IN_DOL_W, IN_DOL_B,
  IN_DO_W, IN_DO_B,
  IN_F_INPROJ_W, IN_F_CONV_W, IN_F_CONV_B, IN_F_XPROJ_W,
  IN_F_DTPROJ_W, IN_F_DTPROJ_B, IN_F_ALOG, IN_F_D, IN_F_OUTPROJ_W,
  IN_B_INPROJ_W, IN_B_CONV_W, IN_B_CONV_B, IN_B_XPROJ_W,
  IN_B_DTPROJ_W, IN_B_DTPROJ_B, IN_B_ALOG, IN_B_D, IN_B_OUTPROJ_W
};

template <int ACT, bool FLIP, bool HASMUL, bool HASRES, bool OUTF32>
static inline void launch_gemm(hipStream_t s, const bf16* A, int lda, const bf16* W,
                               const float* bias, const bf16* mul, const float* res,
                               bf16* outb, float* outf, int N, int K) {
  dim3 grid((N + BN - 1) / BN, MTOT / BM);
  gemm_bf16_kernel<ACT, FLIP, HASMUL, HASRES, OUTF32>
      <<<grid, 256, 0, s>>>(A, lda, W, bias, mul, res, outb, outf, N, K);
}

extern "C" void kernel_launch(void* const* d_in, const int* in_sizes, int n_in,
                              void* d_out, int out_size, void* d_ws, size_t ws_size,
                              hipStream_t stream) {
  (void)in_sizes; (void)n_in; (void)out_size; (void)ws_size;

  const float* hs = (const float*)d_in[IN_HS];

  // ---- workspace carve-out (256B-aligned) ----
  char* base = (char*)d_ws;
  size_t off = 0;
  auto alloc_bf = [&](size_t elems) -> bf16* {
    bf16* p = (bf16*)(base + off);
    off = (off + elems * sizeof(bf16) + 255) & ~(size_t)255;
    return p;
  };

  // bf16 weights
  bf16* dvw   = alloc_bf((size_t)INTER * HIDDEN);
  bf16* dufw  = alloc_bf((size_t)HIDDEN * HIDDEN);
  bf16* dubw  = alloc_bf((size_t)HIDDEN * HIDDEN);
  bf16* ducfw = alloc_bf((size_t)HIDDEN * HIDDEN);
  bf16* ducbw = alloc_bf((size_t)HIDDEN * HIDDEN);
  bf16* dolw  = alloc_bf((size_t)INTER * HIDDEN);
  bf16* dow   = alloc_bf((size_t)HIDDEN * INTER);
  bf16 *ipw[2], *xpw[2], *dtw[2], *opw[2];
  for (int k = 0; k < 2; ++k) {
    ipw[k] = alloc_bf((size_t)2 * D_INNER * HIDDEN);
    xpw[k] = alloc_bf((size_t)(DT_RANK + 2 * D_STATE) * D_INNER);
    dtw[k] = alloc_bf((size_t)D_INNER * DT_RANK);
    opw[k] = alloc_bf((size_t)HIDDEN * D_INNER);
  }
  // activations
  bf16* Xln   = alloc_bf((size_t)MTOT * HIDDEN);
  bf16* Vbuf  = alloc_bf((size_t)MTOT * INTER);
  bf16* U     = alloc_bf((size_t)MTOT * HIDDEN);
  bf16* XZ    = alloc_bf((size_t)MTOT * 2 * D_INNER);
  bf16* XM    = alloc_bf((size_t)MTOT * D_INNER);
  bf16* XDBL  = alloc_bf((size_t)MTOT * 96);
  bf16* DELTA = alloc_bf((size_t)MTOT * D_INNER);
  bf16* YG    = alloc_bf((size_t)MTOT * D_INNER);
  bf16* MO    = alloc_bf((size_t)MTOT * HIDDEN);
  bf16* UCF   = alloc_bf((size_t)MTOT * HIDDEN);
  bf16* PROD  = alloc_bf((size_t)MTOT * HIDDEN);
  bf16* G     = alloc_bf((size_t)MTOT * INTER);

  auto cvt = [&](int idx, bf16* dst, int n) {
    cvt_f32_bf16<<<(n + 255) / 256, 256, 0, stream>>>((const float*)d_in[idx], dst, n);
  };
  cvt(IN_DV_W, dvw, INTER * HIDDEN);
  cvt(IN_DUF_W, dufw, HIDDEN * HIDDEN);
  cvt(IN_DUB_W, dubw, HIDDEN * HIDDEN);
  cvt(IN_DUCF_W, ducfw, HIDDEN * HIDDEN);
  cvt(IN_DUCB_W, ducbw, HIDDEN * HIDDEN);
  cvt(IN_DOL_W, dolw, INTER * HIDDEN);
  cvt(IN_DO_W, dow, HIDDEN * INTER);
  const int fbase[2] = {IN_F_INPROJ_W, IN_B_INPROJ_W};
  for (int k = 0; k < 2; ++k) {
    cvt(fbase[k] + 0, ipw[k], 2 * D_INNER * HIDDEN);
    cvt(fbase[k] + 3, xpw[k], (DT_RANK + 2 * D_STATE) * D_INNER);
    cvt(fbase[k] + 4, dtw[k], D_INNER * DT_RANK);
    cvt(fbase[k] + 8, opw[k], HIDDEN * D_INNER);
  }

  // ---- layernorm ----
  ln_kernel<<<MTOT, 256, 0, stream>>>(hs, (const float*)d_in[IN_LN_G],
                                      (const float*)d_in[IN_LN_B], Xln);

  // ---- v = gelu(dv(hs_ln)) ----
  launch_gemm<ACT_GELU, false, false, false, false>(
      stream, Xln, HIDDEN, dvw, (const float*)d_in[IN_DV_B], nullptr, nullptr,
      Vbuf, nullptr, INTER, HIDDEN);

  auto run_mamba = [&](int k) {
    const float* convw = (const float*)d_in[fbase[k] + 1];
    const float* convb = (const float*)d_in[fbase[k] + 2];
    const float* dtb   = (const float*)d_in[fbase[k] + 5];
    const float* alog  = (const float*)d_in[fbase[k] + 6];
    const float* dvec  = (const float*)d_in[fbase[k] + 7];
    // xz = U @ in_proj^T
    launch_gemm<ACT_NONE, false, false, false, false>(
        stream, U, HIDDEN, ipw[k], nullptr, nullptr, nullptr, XZ, nullptr,
        2 * D_INNER, HIDDEN);
    // depthwise conv + bias + silu
    {
      const int total = BATCH * SEQ * D_INNER;
      conv_silu_kernel<<<(total + 255) / 256, 256, 0, stream>>>(XZ, convw, convb, XM);
    }
    // x_dbl = xm @ x_proj^T  (N = 96)
    launch_gemm<ACT_NONE, false, false, false, false>(
        stream, XM, D_INNER, xpw[k], nullptr, nullptr, nullptr, XDBL, nullptr,
        DT_RANK + 2 * D_STATE, D_INNER);
    // delta = softplus(dt @ dt_proj^T + b)  (A = first 64 cols of x_dbl, lda = 96)
    launch_gemm<ACT_SOFTPLUS, false, false, false, false>(
        stream, XDBL, 96, dtw[k], dtb, nullptr, nullptr, DELTA, nullptr,
        D_INNER, DT_RANK);
    // selective scan + D skip + z gate
    scan_kernel<<<dim3(D_INNER / 256, BATCH), 256, 0, stream>>>(
        DELTA, XM, XDBL, XZ, alog, dvec, YG);
    // mamba out = y @ out_proj^T
    launch_gemm<ACT_NONE, false, false, false, false>(
        stream, YG, D_INNER, opw[k], nullptr, nullptr, nullptr, MO, nullptr,
        HIDDEN, D_INNER);
  };

  // ---- forward branch ----
  launch_gemm<ACT_GELU, false, false, false, false>(
      stream, Xln, HIDDEN, dufw, (const float*)d_in[IN_DUF_B], nullptr, nullptr,
      U, nullptr, HIDDEN, HIDDEN);
  run_mamba(0);
  launch_gemm<ACT_NONE, false, false, false, false>(
      stream, MO, HIDDEN, ducfw, (const float*)d_in[IN_DUCF_B], nullptr, nullptr,
      UCF, nullptr, HIDDEN, HIDDEN);

  // ---- backward branch (flip folded into GEMM row permutation) ----
  launch_gemm<ACT_GELU, true, false, false, false>(
      stream, Xln, HIDDEN, dubw, (const float*)d_in[IN_DUB_B], nullptr, nullptr,
      U, nullptr, HIDDEN, HIDDEN);
  run_mamba(1);
  launch_gemm<ACT_NONE, true, true, false, false>(
      stream, MO, HIDDEN, ducbw, (const float*)d_in[IN_DUCB_B], UCF, nullptr,
      PROD, nullptr, HIDDEN, HIDDEN);   // prod = uc_f * flip(uc_b)

  // ---- combine: g = gelu(dol(prod)) * v ; out = do(g) + hidden_states ----
  launch_gemm<ACT_GELU, false, true, false, false>(
      stream, PROD, HIDDEN, dolw, (const float*)d_in[IN_DOL_B], Vbuf, nullptr,
      G, nullptr, INTER, HIDDEN);
  launch_gemm<ACT_NONE, false, false, true, true>(
      stream, G, INTER, dow, (const float*)d_in[IN_DO_B], nullptr, hs,
      nullptr, (float*)d_out, HIDDEN, INTER);
}
